// Encoderlayer_59390807769596
// MI455X (gfx1250) — matrix operand order repaired and checked by reading, not compiled, not run on hardware
//
#include <hip/hip_runtime.h>

typedef __attribute__((ext_vector_type(16))) _Float16 v16h;
typedef __attribute__((ext_vector_type(8)))  float    v8f;

#define Bz 8
#define Sz 1024
#define Hz 512
#define NHz 16
#define Dz 32

// LDS A-panel row stride (halfs): 512 + 8 pad -> row stride 1040 B (65*16B),
// keeps ds_load_b128 fragment reads conflict-minimal across 64 banks.
#define APAD 520

union Frag16 { v16h h; uint4 u[2]; };
union Pack8h { _Float16 e[8]; uint4 u; };

__device__ inline v8f wmma16(v16h a, v16h b, v8f c) {
  return __builtin_amdgcn_wmma_f32_16x16x32_f16(false, a, false, b, (short)0, c,
                                                false, false);
}

// Stage a contiguous 16-row x 512-col f16 A panel (16 KB) into padded LDS via
// CDNA5 async global->LDS DMA. Caller: 128 threads; waits ASYNCcnt + barrier.
__device__ inline void stage_A_panel(const _Float16* gA, _Float16* As) {
  const char* gbase = (const char*)gA;
  unsigned lbase = (unsigned)(size_t)&As[0];   // low 32 bits = LDS byte offset
#pragma unroll
  for (int t = 0; t < 8; ++t) {
    int c = (int)threadIdx.x + 128 * t;        // 16B chunk id, 0..1023
    unsigned long long ga = (unsigned long long)(size_t)(gbase + c * 16);
    unsigned la = lbase + (unsigned)((c >> 6) * (APAD * 2) + (c & 63) * 16);
    asm volatile("global_load_async_to_lds_b128 %0, %1, off"
                 :: "v"(la), "v"(ga) : "memory");
  }
  asm volatile("s_wait_asynccnt 0x0" ::: "memory");
  __syncthreads();
}

// ---------------- styles = w @ (affine_weight/sqrt(512)).T + affine_bias ----
__global__ void __launch_bounds__(256)
styles_kernel(const float* __restrict__ w, const float* __restrict__ aw,
              const float* __restrict__ ab, float* __restrict__ styles) {
  int t = blockIdx.x * 256 + threadIdx.x;       // 0..8191 -> (b, j)
  int b = t >> 10, j = t & 1023;
  const float inv = 0.044194173824159216f;      // 1/sqrt(512)
  const float* wr = w + b * Hz;
  const float* ar = aw + j * Hz;
  float acc = 0.f;
  for (int i = 0; i < Hz; ++i) acc += wr[i] * ar[i];
  styles[t] = acc * inv + ab[j];
}

// ---------------- demod: rsqrt(sum((W*s)^2)+1e-8); q/k/v use s1, w uses s2 --
__global__ void __launch_bounds__(256)
demod_kernel(const float* __restrict__ qw, const float* __restrict__ kw,
             const float* __restrict__ vw, const float* __restrict__ ww,
             const float* __restrict__ styles, float* __restrict__ dem) {
  int t = blockIdx.x * 256 + threadIdx.x;       // 0..4095 -> (b, o)
  int ty = blockIdx.y;                          // 0..3
  const float* W = (ty == 0) ? qw : (ty == 1) ? kw : (ty == 2) ? vw : ww;
  int b = t >> 9, o = t & 511;
  const float* s = styles + b * 1024 + (ty == 3 ? Hz : 0);
  const float* wr = W + o * Hz;
  float acc = 0.f;
  for (int i = 0; i < Hz; ++i) { float x = wr[i] * s[i]; acc += x * x; }
  dem[ty * (Bz * Hz) + t] = rsqrtf(acc + 1e-8f);
}

// ---------------- f32 -> f16 weight conversion -------------------------------
__global__ void __launch_bounds__(256)
cvt_kernel(const float* __restrict__ src, _Float16* __restrict__ dst, int n) {
  int i = blockIdx.x * 256 + threadIdx.x;
  if (i < n) dst[i] = (_Float16)src[i];
}

// ---------------- modulate + layernorm -> xn (f16) ---------------------------
__global__ void __launch_bounds__(256)
modln_kernel(const float* __restrict__ x, const float* __restrict__ styles,
             _Float16* __restrict__ xn) {
  int wave = threadIdx.x >> 5, lane = threadIdx.x & 31;
  int row = blockIdx.x * 8 + wave;              // 0..8191 = b*1024+s
  int b = row >> 10;
  const float* xr = x + (size_t)row * Hz;
  const float* s1 = styles + b * 1024;
  float v[16], sum = 0.f, sq = 0.f;
#pragma unroll
  for (int i = 0; i < 16; ++i) {
    int c = i * 32 + lane;
    float t = xr[c] * s1[c];
    v[i] = t; sum += t; sq += t * t;
  }
#pragma unroll
  for (int m = 1; m < 32; m <<= 1) {
    sum += __shfl_xor(sum, m, 32);
    sq  += __shfl_xor(sq,  m, 32);
  }
  float mu = sum * (1.f / Hz);
  float var = sq * (1.f / Hz) - mu * mu;
  float rs = rsqrtf(var + 1e-5f);
  _Float16* xo = xn + (size_t)row * Hz;
#pragma unroll
  for (int i = 0; i < 16; ++i) {
    int c = i * 32 + lane;
    xo[c] = (_Float16)((v[i] - mu) * rs);
  }
}

// ---------------- Y = Xn @ W.T  (WMMA), scaled by demod (and s2 for V) -------
// Block = 4 waves sharing one async-staged LDS A panel; each wave: 16x64 tile.
__global__ void __launch_bounds__(128)
gemm_mod_kernel(const _Float16* __restrict__ A, const _Float16* __restrict__ W16,
                const float* __restrict__ dem, const float* __restrict__ s2opt,
                _Float16* __restrict__ out, int transposeV) {
  __shared__ __align__(16) _Float16 As[16 * APAD];
  int wave = threadIdx.x >> 5, lane = threadIdx.x & 31;
  int half = lane >> 4, lm = lane & 15;
  int tileBase = blockIdx.x * 4;                // block-uniform panel
  int bi = tileBase >> 9;                       // batch
  int st = (tileBase & 511) >> 3;               // s-tile 0..63
  int ng = (tileBase & 7) + wave;               // 64-col group 0..7
  int koffA = half * 16;
  int koffB = half * 16;

  stage_A_panel(A + (size_t)(bi * Sz + st * 16) * Hz, As);

  v8f zero = {0.f, 0.f, 0.f, 0.f, 0.f, 0.f, 0.f, 0.f};
  v8f acc[4];
#pragma unroll
  for (int j = 0; j < 4; ++j) acc[j] = zero;

  for (int kk = 0; kk < Hz; kk += 32) {
    Frag16 af;
    af.u[0] = *(const uint4*)(&As[lm * APAD + kk + koffA]);
    af.u[1] = *(const uint4*)(&As[lm * APAD + kk + koffA + 8]);
    Frag16 bf[4];
#pragma unroll
    for (int j = 0; j < 4; ++j) {
      int col = ng * 64 + j * 16 + lm;
      const _Float16* bb = W16 + (size_t)col * Hz + kk + koffB;
      bf[j].u[0] = *(const uint4*)(bb);
      bf[j].u[1] = *(const uint4*)(bb + 8);
      __builtin_prefetch(bb + 32, 0, 3);        // global_prefetch_b8
    }
#pragma unroll
    for (int j = 0; j < 4; ++j) acc[j] = wmma16(af.h, bf[j].h, acc[j]);
  }

#pragma unroll
  for (int j = 0; j < 4; ++j) {
    int col = ng * 64 + j * 16 + lm;
    float sc = dem[bi * Hz + col];
    if (s2opt) sc *= s2opt[bi * 1024 + Hz + col];
    if (!transposeV) {
#pragma unroll
      for (int r = 0; r < 8; ++r) {
        int s = st * 16 + half * 8 + r;
        out[((size_t)(bi * Sz + s)) * Hz + col] = (_Float16)(acc[j][r] * sc);
      }
    } else {
      int h = col >> 5, dm = col & 31;
      Pack8h p;
#pragma unroll
      for (int r = 0; r < 8; ++r) p.e[r] = (_Float16)(acc[j][r] * sc);
      _Float16* dst = out + ((size_t)((bi * NHz + h) * Dz + dm)) * Sz
                          + st * 16 + half * 8;
      *(uint4*)dst = p.u;
    }
  }
}

// ---------------- flash attention: one wave per (b, h, 16-query tile) --------
__global__ void __launch_bounds__(32)
attn_kernel(const _Float16* __restrict__ Q, const _Float16* __restrict__ K,
            const _Float16* __restrict__ Vt, _Float16* __restrict__ O) {
  __shared__ __align__(16) _Float16 pt[16][32];
  int lane = threadIdx.x;
  int half = lane >> 4, lm = lane & 15;
  int st = blockIdx.x, h = blockIdx.y, b = blockIdx.z;
  int koffA = half * 16, koffB = half * 16;
  const float scale = 0.17677669529663687f;     // 1/sqrt(32)

  Frag16 qa;
  const _Float16* qbase = Q + ((size_t)(b * Sz + st * 16 + lm)) * Hz + h * Dz;
  qa.u[0] = *(const uint4*)(qbase + koffA);
  qa.u[1] = *(const uint4*)(qbase + koffA + 8);

  v8f zero = {0.f, 0.f, 0.f, 0.f, 0.f, 0.f, 0.f, 0.f};
  v8f acc0 = zero, acc1 = zero;
  float mi[8], li[8];
#pragma unroll
  for (int r = 0; r < 8; ++r) { mi[r] = -3.0e38f; li[r] = 0.f; }

  const _Float16* vbase = Vt + ((size_t)(b * NHz + h)) * Dz * Sz;

  for (int kb = 0; kb < Sz; kb += 32) {
    Frag16 bk;
    const _Float16* k0 = K + ((size_t)(b * Sz + kb + lm)) * Hz + h * Dz + koffB;
    bk.u[0] = *(const uint4*)(k0);
    bk.u[1] = *(const uint4*)(k0 + 8);
    v8f s0 = wmma16(qa.h, bk.h, zero);
    const _Float16* k1 = K + ((size_t)(b * Sz + kb + 16 + lm)) * Hz + h * Dz + koffB;
    bk.u[0] = *(const uint4*)(k1);
    bk.u[1] = *(const uint4*)(k1 + 8);
    v8f s1 = wmma16(qa.h, bk.h, zero);

#pragma unroll
    for (int r = 0; r < 8; ++r) {
      float a0 = s0[r] * scale, a1 = s1[r] * scale;
      float rm = fmaxf(a0, a1);
#pragma unroll
      for (int m = 1; m < 16; m <<= 1) rm = fmaxf(rm, __shfl_xor(rm, m, 32));
      float mn = fmaxf(mi[r], rm);
      float alpha = __expf(mi[r] - mn);
      float p0 = __expf(a0 - mn), p1 = __expf(a1 - mn);
      float rs = p0 + p1;
#pragma unroll
      for (int m = 1; m < 16; m <<= 1) rs += __shfl_xor(rs, m, 32);
      li[r] = li[r] * alpha + rs;
      mi[r] = mn;
      acc0[r] *= alpha;
      acc1[r] *= alpha;
      int row = half * 8 + r;
      pt[row][lm]      = (_Float16)p0;
      pt[row][16 + lm] = (_Float16)p1;
    }
    __syncthreads();

    Frag16 pa;
    pa.u[0] = *(const uint4*)(&pt[lm][koffA]);
    pa.u[1] = *(const uint4*)(&pt[lm][koffA + 8]);

    Frag16 bv;
    const _Float16* v0 = vbase + (size_t)lm * Sz + kb + koffB;
    bv.u[0] = *(const uint4*)(v0);
    bv.u[1] = *(const uint4*)(v0 + 8);
    acc0 = wmma16(pa.h, bv.h, acc0);
    const _Float16* v1 = vbase + (size_t)(16 + lm) * Sz + kb + koffB;
    bv.u[0] = *(const uint4*)(v1);
    bv.u[1] = *(const uint4*)(v1 + 8);
    acc1 = wmma16(pa.h, bv.h, acc1);
    __syncthreads();
  }

#pragma unroll
  for (int r = 0; r < 8; ++r) {
    float inv = 1.f / li[r];
    int s = st * 16 + half * 8 + r;
    _Float16* ob = O + ((size_t)(b * Sz + s)) * Hz + h * Dz;
    ob[lm]      = (_Float16)(acc0[r] * inv);
    ob[16 + lm] = (_Float16)(acc1[r] * inv);
  }
}

// ---------------- out = LeakyClamp(o @ Ww.T * wd + noise + bias) -------------
__global__ void __launch_bounds__(128)
final_gemm_kernel(const _Float16* __restrict__ A, const _Float16* __restrict__ W16,
                  const float* __restrict__ wd, const float* __restrict__ nco,
                  const float* __restrict__ nstrp, const float* __restrict__ bias,
                  float* __restrict__ out) {
  __shared__ __align__(16) _Float16 As[16 * APAD];
  int wave = threadIdx.x >> 5, lane = threadIdx.x & 31;
  int half = lane >> 4, lm = lane & 15;
  int tileBase = blockIdx.x * 4;
  int bi = tileBase >> 9;
  int st = (tileBase & 511) >> 3;
  int ng = (tileBase & 7) + wave;
  int koffA = half * 16;
  int koffB = half * 16;

  stage_A_panel(A + (size_t)(bi * Sz + st * 16) * Hz, As);

  v8f zero = {0.f, 0.f, 0.f, 0.f, 0.f, 0.f, 0.f, 0.f};
  v8f acc[4];
#pragma unroll
  for (int j = 0; j < 4; ++j) acc[j] = zero;

  for (int kk = 0; kk < Hz; kk += 32) {
    Frag16 af;
    af.u[0] = *(const uint4*)(&As[lm * APAD + kk + koffA]);
    af.u[1] = *(const uint4*)(&As[lm * APAD + kk + koffA + 8]);
    Frag16 bf[4];
#pragma unroll
    for (int j = 0; j < 4; ++j) {
      int col = ng * 64 + j * 16 + lm;
      const _Float16* bb = W16 + (size_t)col * Hz + kk + koffB;
      bf[j].u[0] = *(const uint4*)(bb);
      bf[j].u[1] = *(const uint4*)(bb + 8);
      __builtin_prefetch(bb + 32, 0, 3);
    }
#pragma unroll
    for (int j = 0; j < 4; ++j) acc[j] = wmma16(af.h, bf[j].h, acc[j]);
  }

  float nstr = *nstrp;
#pragma unroll
  for (int j = 0; j < 4; ++j) {
    int col = ng * 64 + j * 16 + lm;
    float sc = wd[bi * Hz + col];
    float bv = bias[col];
#pragma unroll
    for (int r = 0; r < 8; ++r) {
      int s = st * 16 + half * 8 + r;
      float v = acc[j][r] * sc + nco[s] * nstr + bv;
      v = (v >= 0.f) ? v : 0.2f * v;
      v = fminf(fmaxf(v, -256.f), 256.f);
      out[((size_t)(bi * Sz + s)) * Hz + col] = v;
    }
  }
}

// ---------------------------------------------------------------------------
extern "C" void kernel_launch(void* const* d_in, const int* in_sizes, int n_in,
                              void* d_out, int out_size, void* d_ws, size_t ws_size,
                              hipStream_t stream) {
  (void)in_sizes; (void)n_in; (void)out_size; (void)ws_size;
  const float* x    = (const float*)d_in[0];
  const float* w    = (const float*)d_in[1];
  const float* aw   = (const float*)d_in[2];
  const float* ab   = (const float*)d_in[3];
  const float* qw   = (const float*)d_in[4];
  const float* kw   = (const float*)d_in[5];
  const float* vw   = (const float*)d_in[6];
  const float* wwgt = (const float*)d_in[7];
  // d_in[8] = u_weight: unused by the reference
  const float* nco  = (const float*)d_in[9];
  const float* nstr = (const float*)d_in[10];
  const float* bias = (const float*)d_in[11];

  char* ws = (char*)d_ws;
  size_t off = 0;
  auto alloc = [&](size_t bytes) -> void* {
    void* p = ws + off;
    off += (bytes + 255) & ~(size_t)255;
    return p;
  };
  float*     styles = (float*)    alloc((size_t)Bz * 1024 * 4);
  float*     dem    = (float*)    alloc((size_t)4 * Bz * Hz * 4);
  _Float16*  wq16   = (_Float16*) alloc((size_t)Hz * Hz * 2);
  _Float16*  wk16   = (_Float16*) alloc((size_t)Hz * Hz * 2);
  _Float16*  wv16   = (_Float16*) alloc((size_t)Hz * Hz * 2);
  _Float16*  ww16   = (_Float16*) alloc((size_t)Hz * Hz * 2);
  _Float16*  xn16   = (_Float16*) alloc((size_t)Bz * Sz * Hz * 2);
  _Float16*  q16    = (_Float16*) alloc((size_t)Bz * Sz * Hz * 2);
  _Float16*  k16    = (_Float16*) alloc((size_t)Bz * Sz * Hz * 2);
  _Float16*  vt16   = (_Float16*) alloc((size_t)Bz * Sz * Hz * 2);
  _Float16*  o16    = xn16;   // xn16 dead after QKV GEMMs -> reuse for attn out

  const int WN = Hz * Hz; // 262144 per weight

  styles_kernel<<<32, 256, 0, stream>>>(w, aw, ab, styles);
  demod_kernel<<<dim3(16, 4), 256, 0, stream>>>(qw, kw, vw, wwgt, styles, dem);
  cvt_kernel<<<(WN + 255) / 256, 256, 0, stream>>>(qw,   wq16, WN);
  cvt_kernel<<<(WN + 255) / 256, 256, 0, stream>>>(kw,   wk16, WN);
  cvt_kernel<<<(WN + 255) / 256, 256, 0, stream>>>(vw,   wv16, WN);
  cvt_kernel<<<(WN + 255) / 256, 256, 0, stream>>>(wwgt, ww16, WN);
  modln_kernel<<<1024, 256, 0, stream>>>(x, styles, xn16);

  // QKV projections (4096 wave-tiles of 16x64 each; 4 waves/block, LDS A panel)
  gemm_mod_kernel<<<1024, 128, 0, stream>>>(xn16, wq16, dem + 0 * Bz * Hz,
                                            nullptr, q16, 0);
  gemm_mod_kernel<<<1024, 128, 0, stream>>>(xn16, wk16, dem + 1 * Bz * Hz,
                                            nullptr, k16, 0);
  gemm_mod_kernel<<<1024, 128, 0, stream>>>(xn16, wv16, dem + 2 * Bz * Hz,
                                            styles, vt16, 1);

  attn_kernel<<<dim3(Sz / 16, NHz, Bz), 32, 0, stream>>>(q16, k16, vt16, o16);

  final_gemm_kernel<<<1024, 128, 0, stream>>>(o16, ww16, dem + 3 * Bz * Hz,
                                              nco, nstr, bias, (float*)d_out);
}
